// MultiLayerRGCN_48773648613822
// MI455X (gfx1250) — matrix operand order repaired and checked by reading, not compiled, not run on hardware
//
#include <hip/hip_runtime.h>

typedef __attribute__((ext_vector_type(16))) _Float16 v16h;
typedef __attribute__((ext_vector_type(8)))  float    v8f;

#define NREL 8

// ---------------- zero (float4 stores) ----------------
__global__ __launch_bounds__(256) void k_zero4(float4* __restrict__ p, size_t n4) {
  size_t i = (size_t)blockIdx.x * blockDim.x + threadIdx.x;
  if (i < n4) p[i] = make_float4(0.f, 0.f, 0.f, 0.f);
}

// ---------------- edge scatter: agg[(dst*R+rel)*D + c] += feat[src*D + c] ----------------
template <int D>
__global__ __launch_bounds__(256) void k_scatter(
    const float* __restrict__ feat, const int* __restrict__ src,
    const int* __restrict__ dst, const int* __restrict__ rel,
    float* __restrict__ agg, int E) {
  constexpr int C = D / 4;  // float4 chunks per edge
  size_t t = (size_t)blockIdx.x * blockDim.x + threadIdx.x;
  if (t >= (size_t)E * C) return;
  int e = (int)(t / C);
  int c = (int)(t % C) * 4;
  float4 v = *reinterpret_cast<const float4*>(feat + (size_t)src[e] * D + c);
  float* base = agg + ((size_t)dst[e] * NREL + rel[e]) * D + c;
  atomicAdd(base + 0, v.x);
  atomicAdd(base + 1, v.y);
  atomicAdd(base + 2, v.z);
  atomicAdd(base + 3, v.w);
}

// ---------------- per-(node,relation) edge counts ----------------
__global__ __launch_bounds__(256) void k_count(
    const int* __restrict__ dst, const int* __restrict__ rel,
    float* __restrict__ cnt, int E) {
  size_t t = (size_t)blockIdx.x * blockDim.x + threadIdx.x;
  if (t < (size_t)E) atomicAdd(&cnt[(size_t)dst[t] * NREL + rel[t]], 1.0f);
}

// ---------------- inv = 1 / max(cnt, 1) ----------------
__global__ __launch_bounds__(256) void k_inv(const float* __restrict__ cnt,
                                             float* __restrict__ inv, int n) {
  int i = blockIdx.x * blockDim.x + threadIdx.x;
  if (i < n) inv[i] = 1.0f / fmaxf(cnt[i], 1.0f);
}

// ---------------- pack [W_rel (R*D x H); W_root (D x H)] -> f16 row-major ----------------
template <int D, int H>
__global__ __launch_bounds__(256) void k_packB(const float* __restrict__ Wrel,
                                               const float* __restrict__ Wroot,
                                               _Float16* __restrict__ Bp) {
  constexpr int KA = NREL * D;
  constexpr size_t TOT = (size_t)(KA + D) * H;
  size_t i = (size_t)blockIdx.x * blockDim.x + threadIdx.x;
  if (i >= TOT) return;
  int k = (int)(i / H);
  int h = (int)(i % H);
  float v = (k < KA) ? Wrel[i] : Wroot[(size_t)(k - KA) * H + h];
  Bp[i] = (_Float16)v;
}

// float4 -> 4 scaled halves into a v16h slot
__device__ __forceinline__ void cvt4s(v16h& a, int base, float4 f, float s) {
  a[base + 0] = (_Float16)(f.x * s);
  a[base + 1] = (_Float16)(f.y * s);
  a[base + 2] = (_Float16)(f.z * s);
  a[base + 3] = (_Float16)(f.w * s);
}
__device__ __forceinline__ void cvt4(v16h& a, int base, float4 f) {
  a[base + 0] = (_Float16)f.x;
  a[base + 1] = (_Float16)f.y;
  a[base + 2] = (_Float16)f.z;
  a[base + 3] = (_Float16)f.w;
}

// ---------------- fused RGCN GEMM: out = relu([agg/cnt | feat] @ Bp + bias) ----------------
// One wave per 16x16 output tile; K stepped by 32 with v_wmma_f32_16x16x32_f16.
// Branch-free: relation region (scaled) and root region (unscaled) are separate loops.
template <int D, int H>
__global__ __launch_bounds__(256) void k_gemm(
    const float* __restrict__ agg, const float* __restrict__ invc,
    const float* __restrict__ feat, const _Float16* __restrict__ Bp,
    const float* __restrict__ bias, float* __restrict__ out) {
  constexpr int KA = NREL * D;   // aggregated-feature K region
  constexpr int NT = H / 16;     // column tiles (power of two -> shifts)

  int wid  = blockIdx.x * (blockDim.x >> 5) + (threadIdx.x >> 5);
  int lane = threadIdx.x & 31;
  int tm   = wid / NT;
  int tn   = wid % NT;
  int m    = lane & 15;          // A row within tile / C column index
  int half = lane >> 4;          // K-half selector for A, M-half for C
  int row  = tm * 16 + m;        // node index
  int h0   = tn * 16;            // output-feature tile base

  // lane-private base pointers; each K-step reads 8 floats at +0 and 8 at +16
  const float*    ap = agg  + (size_t)row * KA + half * 8;
  const float*    fp = feat + (size_t)row * D  + half * 8;
  const float*    irow = invc + (size_t)row * NREL;
  const _Float16* bp = Bp + (size_t)(half * 8) * H + h0 + m;  // B col h0+m, rows k+8*half+{0..7,16..23}

  v8f acc = {};

  // ---- aggregated-relation region: scale hoisted per relation (32 | D) ----
  for (int r = 0; r < NREL; ++r) {
    float s = irow[r];
    const float* pr = ap + r * D;
#pragma unroll
    for (int kk = 0; kk < D; kk += 32) {
      const float* p = pr + kk;
      __builtin_prefetch(p + 256, 0, 1);  // speculative; OOB prefetches are dropped
      float4 f0 = *reinterpret_cast<const float4*>(p);
      float4 f1 = *reinterpret_cast<const float4*>(p + 4);
      float4 f2 = *reinterpret_cast<const float4*>(p + 16);
      float4 f3 = *reinterpret_cast<const float4*>(p + 20);
      v16h a;
      cvt4s(a, 0, f0, s);
      cvt4s(a, 4, f1, s);
      cvt4s(a, 8, f2, s);
      cvt4s(a, 12, f3, s);
      v16h b;
      for (int s = 0; s < 16; ++s) b[s] = bp[(size_t)(r * D + kk + s + (s >= 8 ? 8 : 0)) * H];
      acc = __builtin_amdgcn_wmma_f32_16x16x32_f16(false, a, false, b, (short)0, acc,
                                                   false, false);
    }
  }

  // ---- root-feature region: no scaling ----
#pragma unroll
  for (int kk = 0; kk < D; kk += 32) {
    const float* p = fp + kk;
    float4 f0 = *reinterpret_cast<const float4*>(p);
    float4 f1 = *reinterpret_cast<const float4*>(p + 4);
    float4 f2 = *reinterpret_cast<const float4*>(p + 16);
    float4 f3 = *reinterpret_cast<const float4*>(p + 20);
    v16h a;
    cvt4(a, 0, f0);
    cvt4(a, 4, f1);
    cvt4(a, 8, f2);
    cvt4(a, 12, f3);
    v16h b;
    for (int s = 0; s < 16; ++s) b[s] = bp[(size_t)(KA + kk + s + (s >= 8 ? 8 : 0)) * H];
    acc = __builtin_amdgcn_wmma_f32_16x16x32_f16(false, a, false, b, (short)0, acc,
                                                 false, false);
  }

  // ---- C/D: VGPR v -> M = v + 8*half, N = lane&15. Fuse bias + ReLU. ----
  int col = h0 + m;
  float bv = bias[col];
  float* orow = out + (size_t)(tm * 16 + half * 8) * H + col;
#pragma unroll
  for (int v = 0; v < 8; ++v) {
    orow[(size_t)v * H] = fmaxf(acc[v] + bv, 0.0f);
  }
}

extern "C" void kernel_launch(void* const* d_in, const int* in_sizes, int n_in,
                              void* d_out, int out_size, void* d_ws, size_t ws_size,
                              hipStream_t stream) {
  const float* x      = (const float*)d_in[0];
  const int*   eidx   = (const int*)  d_in[1];   // [2 layers][src,dst][E]
  const int*   etyp   = (const int*)  d_in[2];   // [2 layers][E]
  const float* Wrel0  = (const float*)d_in[3];
  const float* Wroot0 = (const float*)d_in[4];
  const float* b0     = (const float*)d_in[5];
  const float* Wrel1  = (const float*)d_in[6];
  const float* Wroot1 = (const float*)d_in[7];
  const float* b1     = (const float*)d_in[8];
  float* out = (float*)d_out;

  constexpr int D0 = 128, DH = 256, HOUT = 128;
  const int N = in_sizes[0] / D0;      // 50000
  const int E = in_sizes[2] / 2;       // 1600000

  // ---- workspace carve-out (256B aligned) ----
  char* ws = (char*)d_ws;
  size_t off = 0;
  auto alloc = [&](size_t bytes) -> void* {
    void* p = ws + off;
    off = (off + bytes + 255) & ~(size_t)255;
    return p;
  };
  float*    agg  = (float*)   alloc((size_t)N * NREL * DH * sizeof(float)); // reused both layers
  float*    cnt  = (float*)   alloc((size_t)N * NREL * sizeof(float));
  float*    inv  = (float*)   alloc((size_t)N * NREL * sizeof(float));
  float*    hbuf = (float*)   alloc((size_t)N * DH * sizeof(float));
  _Float16* Bp   = (_Float16*)alloc((size_t)(NREL + 1) * DH * 256 * sizeof(_Float16));
  (void)ws_size; (void)n_in; (void)out_size;

  const int TB = 256;
  auto nb = [](size_t n, int tb) { return (unsigned)((n + (size_t)tb - 1) / tb); };

  const int* src0 = eidx;
  const int* dst0 = eidx + (size_t)E;
  const int* src1 = eidx + 2 * (size_t)E;
  const int* dst1 = eidx + 3 * (size_t)E;
  const int* rel0 = etyp;
  const int* rel1 = etyp + (size_t)E;

  const size_t cF = (size_t)N * NREL;  // 400000, divisible by 4

  // ================= Layer 0: x[N,128] -> hbuf[N,256] =================
  {
    size_t aggF = (size_t)N * NREL * D0;
    k_zero4<<<nb(aggF / 4, TB), TB, 0, stream>>>((float4*)agg, aggF / 4);
    k_zero4<<<nb(cF / 4, TB), TB, 0, stream>>>((float4*)cnt, cF / 4);
    k_scatter<D0><<<nb((size_t)E * (D0 / 4), TB), TB, 0, stream>>>(x, src0, dst0, rel0, agg, E);
    k_count<<<nb((size_t)E, TB), TB, 0, stream>>>(dst0, rel0, cnt, E);
    k_inv<<<nb(cF, TB), TB, 0, stream>>>(cnt, inv, (int)cF);
    k_packB<D0, DH><<<nb((size_t)(NREL + 1) * D0 * DH, TB), TB, 0, stream>>>(Wrel0, Wroot0, Bp);
    int tiles = (N / 16) * (DH / 16);   // 50000 tiles, 8 waves per block
    k_gemm<D0, DH><<<tiles / 8, TB, 0, stream>>>(agg, inv, x, Bp, b0, hbuf);
  }

  // ================= Layer 1: hbuf[N,256] -> out[N,128] =================
  {
    size_t aggF = (size_t)N * NREL * DH;
    k_zero4<<<nb(aggF / 4, TB), TB, 0, stream>>>((float4*)agg, aggF / 4);
    k_zero4<<<nb(cF / 4, TB), TB, 0, stream>>>((float4*)cnt, cF / 4);
    k_scatter<DH><<<nb((size_t)E * (DH / 4), TB), TB, 0, stream>>>(hbuf, src1, dst1, rel1, agg, E);
    k_count<<<nb((size_t)E, TB), TB, 0, stream>>>(dst1, rel1, cnt, E);
    k_inv<<<nb(cF, TB), TB, 0, stream>>>(cnt, inv, (int)cF);
    k_packB<DH, HOUT><<<nb((size_t)(NREL + 1) * DH * HOUT, TB), TB, 0, stream>>>(Wrel1, Wroot1, Bp);
    int tiles = (N / 16) * (HOUT / 16);  // 25000 tiles, 8 waves per block
    k_gemm<DH, HOUT><<<tiles / 8, TB, 0, stream>>>(agg, inv, hbuf, Bp, b1, out);
  }
}
